// DeformableConvV2_27900107555312
// MI455X (gfx1250) — matrix operand order repaired and checked by reading, not compiled, not run on hardware
//
#include <hip/hip_runtime.h>

// ---------------------------------------------------------------------------
// Deformable Conv v2 for MI455X (gfx1250, wave32, WMMA).
// Both convolutions are lowered to v_wmma_f32_16x16x32_f16 GEMMs with f32
// accumulation; bilinear sampling writes the deformed im2col in f16 directly
// in the WMMA B-matrix-friendly layout ([n][k], k contiguous).
// Main GEMM stages B tiles into LDS via GLOBAL_LOAD_ASYNC_TO_LDS_B128
// (ASYNCcnt) — builtin takes (global int4*, local int4*, imm off, imm cpol).
// Workspace (~45 MB, L2-resident on the 192MB L2):
//   colbuf f16 [9216][2304]  (im2col0, then overwritten by deformed patches)
//   a_dcn  f16 [256][2304], a_off f16 [32][2304], om f32 [32][9216]
// ---------------------------------------------------------------------------

typedef __attribute__((ext_vector_type(16))) _Float16 v16h;
typedef __attribute__((ext_vector_type(8)))  _Float16 v8h;
typedef __attribute__((ext_vector_type(8)))  float    v8f;
typedef __attribute__((ext_vector_type(4)))  int      v4i;

union V16 { v16h v; v8h h[2]; };

typedef __attribute__((address_space(1))) v4i gv4i;   // global int4
typedef __attribute__((address_space(3))) v4i lv4i;   // LDS int4

#if defined(__has_builtin)
#if __has_builtin(__builtin_amdgcn_global_load_async_to_lds_b128)
#define HAVE_ASYNC_LDS 1
#endif
#if __has_builtin(__builtin_amdgcn_s_wait_asynccnt)
#define HAVE_WAIT_ASYNC 1
#endif
#endif

constexpr int BATCH = 4, CIN = 256, COUT = 256, H = 48, W = 48, K2 = 9;
constexpr int KDIM = CIN * K2;        // 2304 (GEMM K)
constexpr int HW   = H * W;           // 2304
constexpr int NTOT = BATCH * HW;      // 9216 (GEMM N)
constexpr int OFF_M = 32;             // 27 offset/mask channels padded to 32

// workspace byte offsets
constexpr long WS_COLBUF = 0;                                   // 42,467,328 B
constexpr long WS_ADCN   = (long)NTOT * KDIM * 2;               // +1,179,648 B
constexpr long WS_AOFF   = WS_ADCN + (long)COUT * KDIM * 2;     // +  147,456 B
constexpr long WS_OM     = WS_AOFF + (long)OFF_M * KDIM * 2;    // +1,179,648 B

__device__ __forceinline__ void stage_b128(const _Float16* g, _Float16* l) {
#ifdef HAVE_ASYNC_LDS
  __builtin_amdgcn_global_load_async_to_lds_b128(
      (gv4i*)const_cast<_Float16*>(g), (lv4i*)l, 0, 0);
#else
  *(v8h*)l = *(const v8h*)g;
#endif
}

__device__ __forceinline__ void wait_stage() {
#ifdef HAVE_ASYNC_LDS
#ifdef HAVE_WAIT_ASYNC
  __builtin_amdgcn_s_wait_asynccnt(0);
#else
  asm volatile("s_wait_asynccnt 0x0" ::: "memory");
#endif
#endif
}

// ---- 1) weight conversion to f16 ------------------------------------------
__global__ __launch_bounds__(256) void k_convert(const float* __restrict__ w_off,
                                                 const float* __restrict__ w_dcn,
                                                 _Float16* __restrict__ a_off,
                                                 _Float16* __restrict__ a_dcn) {
  int i = blockIdx.x * 256 + threadIdx.x;
  if (i < OFF_M * KDIM) {
    int r = i / KDIM, kk = i - r * KDIM;
    a_off[i] = (_Float16)(r < 27 ? w_off[r * KDIM + kk] : 0.f);
  }
  if (i < COUT * KDIM) a_dcn[i] = (_Float16)w_dcn[i];
}

// ---- 2) regular im2col (zero-padded 3x3) to f16, layout [n][c*9+k2] --------
__global__ __launch_bounds__(256) void k_im2col(const float* __restrict__ x,
                                                _Float16* __restrict__ cb) {
  int idx = blockIdx.x * 256 + threadIdx.x;          // < 9216*2304
  int n  = idx / KDIM, kk = idx - n * KDIM;
  int c  = kk / K2,    k2 = kk - c * K2;
  int ki = k2 / 3,     kj = k2 - ki * 3;
  int b  = n / HW,     hw = n - b * HW;
  int h  = hw / W,     w  = hw - h * W;
  int yy = h - 1 + ki, xx = w - 1 + kj;
  float v = 0.f;
  if ((unsigned)yy < (unsigned)H && (unsigned)xx < (unsigned)W)
    v = x[((long)(b * CIN + c) * H + yy) * W + xx];
  cb[idx] = (_Float16)v;
}

// ---- 3) offset/mask conv as WMMA GEMM: om[32x9216] = a_off x colbuf^T ------
// 1 block = 2 waves; wave computes a 16(M) x 64(N) tile, B loaded from global.
__global__ __launch_bounds__(64) void k_gemm_off(const _Float16* __restrict__ A,
                                                 const _Float16* __restrict__ Bm,
                                                 float* __restrict__ om) {
  const int n0   = blockIdx.x * 64;
  const int lane = threadIdx.x & 31;
  const int mt   = threadIdx.x >> 5;      // 0..1 -> M tile
  const int col  = lane & 15;
  const int hi   = lane >> 4;             // 0/1 (half-wave)
  const int klo  = hi * 8;                // unused; A loads with koff too
  const int koff = hi * 16;               // B: K sub-offset
  const int mrow = mt * 16 + col;

  v8f acc[4] = {};
  for (int k0 = 0; k0 < KDIM; k0 += 32) {
    V16 a;
    const _Float16* ap = A + (long)mrow * KDIM + k0 + koff;
    a.h[0] = *(const v8h*)(ap);
    a.h[1] = *(const v8h*)(ap + 8);
#pragma unroll
    for (int nt = 0; nt < 4; ++nt) {
      int n = n0 + nt * 16 + col;
      v16h b = *(const v16h*)(Bm + (long)n * KDIM + k0 + koff);
      acc[nt] = __builtin_amdgcn_wmma_f32_16x16x32_f16(
          false, a.v, false, b, (short)0, acc[nt], false, false);
    }
  }
  const int rbase = hi * 8;
#pragma unroll
  for (int nt = 0; nt < 4; ++nt) {
    int n = n0 + nt * 16 + col;
#pragma unroll
    for (int v = 0; v < 8; ++v)
      om[(long)(mt * 16 + rbase + v) * NTOT + n] = acc[nt][v];
  }
}

// ---- 4) deformed bilinear sampling * sigmoid(mask) -> f16 im2col -----------
__global__ __launch_bounds__(256) void k_sample(const float* __restrict__ x,
                                                const float* __restrict__ om,
                                                _Float16* __restrict__ Dm) {
  const int n = blockIdx.x;
  const int b = n / HW, hw = n - b * HW;
  const int h = hw / W, w  = hw - h * W;
  __shared__ int   sy0[K2], sx0[K2];
  __shared__ float sfy[K2], sfx[K2], sm[K2];
  const int t = threadIdx.x;
  if (t < K2) {
    int ki = t / 3, kj = t - ki * 3;
    float dy = om[(2 * t)     * NTOT + n];
    float dx = om[(2 * t + 1) * NTOT + n];
    float mv = om[(18 + t)    * NTOT + n];
    float py = dy + (float)(h - 1 + ki);
    float px = dx + (float)(w - 1 + kj);
    float fy = floorf(py), fx = floorf(px);
    sy0[t] = (int)fy; sx0[t] = (int)fx;
    sfy[t] = py - fy; sfx[t] = px - fx;
    sm[t]  = 1.f / (1.f + __expf(-mv));
  }
  __syncthreads();
  const int c = t;
  const float* xb = x + (long)(b * CIN + c) * HW;
  _Float16* dp = Dm + (long)n * KDIM + c * K2;
#pragma unroll
  for (int k2 = 0; k2 < K2; ++k2) {
    int   y0 = sy0[k2], x0 = sx0[k2];
    float wy = sfy[k2], wx = sfx[k2];
    float v = 0.f;
    if ((unsigned)y0       < (unsigned)H && (unsigned)x0       < (unsigned)W) v += (1.f-wy)*(1.f-wx)*xb[y0*W + x0];
    if ((unsigned)y0       < (unsigned)H && (unsigned)(x0 + 1) < (unsigned)W) v += (1.f-wy)*wx      *xb[y0*W + x0 + 1];
    if ((unsigned)(y0 + 1) < (unsigned)H && (unsigned)x0       < (unsigned)W) v += wy*(1.f-wx)      *xb[(y0+1)*W + x0];
    if ((unsigned)(y0 + 1) < (unsigned)H && (unsigned)(x0 + 1) < (unsigned)W) v += wy*wx            *xb[(y0+1)*W + x0 + 1];
    dp[k2] = (_Float16)(v * sm[k2]);
  }
}

// ---- 5) main WMMA GEMM: out[256x9216] = a_dcn x D^T ------------------------
// block = 4 waves computing a 128(M) x 64(N) tile; each wave owns 32x64
// (8 accumulators, 2 A fragments share every B fragment). B tile staged in
// LDS ([64 n][32 k] halves) with async global->LDS copies when available.
__global__ __launch_bounds__(128) void k_gemm_main(const _Float16* __restrict__ A,
                                                   const _Float16* __restrict__ Bm,
                                                   float* __restrict__ out) {
  __shared__ __align__(64) _Float16 sB[64 * 32];
  const int n0   = blockIdx.x * 64;
  const int m0   = blockIdx.y * 128;
  const int t    = threadIdx.x;
  const int lane = t & 31;
  const int wm   = t >> 5;               // wave -> 32-row M sub-tile
  const int col  = lane & 15;
  const int hi   = lane >> 4;
  const int klo  = hi * 8;
  const int koff = hi * 16;
  const int mrow0 = m0 + wm * 32 + col;
  const int mrow1 = mrow0 + 16;
  // staging: each of 128 threads copies 16 halves (32 B) of the B tile
  const int snn = t >> 1;
  const int skk = (t & 1) * 16;
  const long srow = (long)(n0 + snn) * KDIM;

  v8f acc0[4] = {}, acc1[4] = {};
  for (int k0 = 0; k0 < KDIM; k0 += 32) {
    __syncthreads();
    stage_b128(&Bm[srow + k0 + skk],     &sB[snn * 32 + skk]);
    stage_b128(&Bm[srow + k0 + skk + 8], &sB[snn * 32 + skk + 8]);
    wait_stage();
    __syncthreads();
    V16 a0, a1;
    const _Float16* ap0 = A + (long)mrow0 * KDIM + k0 + koff;
    const _Float16* ap1 = A + (long)mrow1 * KDIM + k0 + koff;
    a0.h[0] = *(const v8h*)(ap0);
    a0.h[1] = *(const v8h*)(ap0 + 8);
    a1.h[0] = *(const v8h*)(ap1);
    a1.h[1] = *(const v8h*)(ap1 + 8);
#pragma unroll
    for (int nt = 0; nt < 4; ++nt) {
      v16h b = *(const v16h*)&sB[(nt * 16 + col) * 32 + koff];
      acc0[nt] = __builtin_amdgcn_wmma_f32_16x16x32_f16(
          false, a0.v, false, b, (short)0, acc0[nt], false, false);
      acc1[nt] = __builtin_amdgcn_wmma_f32_16x16x32_f16(
          false, a1.v, false, b, (short)0, acc1[nt], false, false);
    }
  }
  const int rbase = hi * 8;
#pragma unroll
  for (int nt = 0; nt < 4; ++nt) {
    int n = n0 + nt * 16 + col;
    int b = n / HW, hw = n - b * HW;
    float* op = out + (long)b * COUT * HW + hw;
#pragma unroll
    for (int v = 0; v < 8; ++v) {
      op[(long)(mrow0 - col + rbase + v) * HW] = acc0[nt][v];
      op[(long)(mrow1 - col + rbase + v) * HW] = acc1[nt][v];
    }
  }
}

// ---------------------------------------------------------------------------
extern "C" void kernel_launch(void* const* d_in, const int* in_sizes, int n_in,
                              void* d_out, int out_size, void* d_ws, size_t ws_size,
                              hipStream_t stream) {
  (void)in_sizes; (void)n_in; (void)out_size; (void)ws_size;
  const float* x     = (const float*)d_in[0];   // [4,256,48,48]
  const float* w_off = (const float*)d_in[1];   // [27,256,3,3]
  const float* w_dcn = (const float*)d_in[2];   // [256,256,3,3]
  float* out = (float*)d_out;                   // [4,256,48,48]

  char* ws = (char*)d_ws;
  _Float16* colbuf = (_Float16*)(ws + WS_COLBUF);
  _Float16* a_dcn  = (_Float16*)(ws + WS_ADCN);
  _Float16* a_off  = (_Float16*)(ws + WS_AOFF);
  float*    om     = (float*)(ws + WS_OM);

  k_convert<<<(COUT * KDIM + 255) / 256, 256, 0, stream>>>(w_off, w_dcn, a_off, a_dcn);
  k_im2col<<<(NTOT * KDIM) / 256, 256, 0, stream>>>(x, colbuf);
  k_gemm_off<<<NTOT / 64, 64, 0, stream>>>(a_off, colbuf, om);
  k_sample<<<NTOT, 256, 0, stream>>>(x, om, colbuf);   // overwrites colbuf
  dim3 g(NTOT / 64, COUT / 128);
  k_gemm_main<<<g, 128, 0, stream>>>(a_dcn, colbuf, out);
}
